// FactsConverterWithQuery_18322330485081
// MI455X (gfx1250) — matrix operand order repaired and checked by reading, not compiled, not run on hardware
//
#include <hip/hip_runtime.h>
#include <hip/hip_bf16.h>

typedef __attribute__((ext_vector_type(16))) _Float16 v16h;
typedef __attribute__((ext_vector_type(8)))  float    v8f;
typedef __attribute__((ext_vector_type(4)))  float    v4f;

#define NB    64    // batch B
#define DD    64    // Z feature dim D
#define QDIM  64    // query dim
#define KDIM  128   // D + QDIM

// ---------------------------------------------------------------------------
// Kernel 1: build AT[k][b] (f16, K-major transposed A matrix) in workspace.
//   k <  64 : AT[k][b] = sum_e Z[b,e,k]
//   k >= 64 : AT[k][b] = Q[b][k-64]
// 8192 elements -> 16 KB, stays hot in L2 for the GEMM's B-fragment loads.
// ---------------------------------------------------------------------------
__global__ void prep_at_kernel(const float* __restrict__ Z,
                               const float* __restrict__ Q,
                               _Float16* __restrict__ AT, int E) {
    int t = blockIdx.x * blockDim.x + threadIdx.x;
    if (t >= KDIM * NB) return;
    int b = t & (NB - 1);
    int k = t >> 6;
    float v;
    if (k < DD) {
        const float* zp = Z + (size_t)b * E * DD + k;
        float s = 0.0f;
        for (int e = 0; e < E; ++e) s += zp[(size_t)e * DD];
        v = s;
    } else {
        v = Q[(size_t)b * QDIM + (k - DD)];
    }
    AT[(size_t)k * NB + b] = (_Float16)v;   // consecutive b -> coalesced b16 writes
}

// ---------------------------------------------------------------------------
// Kernel 2: zero V[:, n_np : n_atoms] with float4 stores.
// ---------------------------------------------------------------------------
__global__ void zero_tail_kernel(float* __restrict__ V, int n_np, int n_atoms) {
    int q = (n_atoms - n_np) >> 2;              // float4 per row (tail % 4 == 0)
    long total = (long)NB * q;
    long stride = (long)gridDim.x * blockDim.x;
    for (long t = (long)blockIdx.x * blockDim.x + threadIdx.x; t < total; t += stride) {
        int b = (int)(t / q);
        int o = (int)(t % q);
        v4f z = {0.f, 0.f, 0.f, 0.f};
        *(v4f*)(V + (size_t)b * n_atoms + n_np + (size_t)o * 4) = z;
    }
}

// ---------------------------------------------------------------------------
// Kernel 3: WMMA GEMM + sigmoid + scatter store.
// One wave handles a 16-row n-tile of [W|U] against all 64 batch columns.
//   A-operand (streamed): 16x32 f16 tile of Bm rows, laid out per ISA:
//     lane<16  : row n0+lane,      halves 0..7 <- K c..c+7, 8..15 <- K c+16..c+23 (c = kk*32)
//     lane>=16 : row n0+lane-16,   halves 0..7 <- K c+8..c+15, 8..15 <- K c+24..c+31
//   B-operand (L2-hot): lane holds b col bt*16+lane%16, halves follow the same
//     K order as the A-operand (strided gather from AT).
// D layout: lane<16 -> b col = bt*16+lane, VGPR j -> n row n0+j;
//           lane>=16 -> rows n0+8..15. Scatter through np_indices.
// ---------------------------------------------------------------------------
__global__ void __launch_bounds__(128)
gemm_sigmoid_kernel(const float* __restrict__ W, const float* __restrict__ U,
                    const _Float16* __restrict__ AT,
                    const int* __restrict__ npidx,
                    float* __restrict__ V, int n_np, int n_atoms) {
    int lane = threadIdx.x & 31;
    int wave = threadIdx.x >> 5;
    int tile = blockIdx.x * 4 + wave;
    int n0 = tile * 16;
    if (n0 >= n_np) return;                    // wave-uniform; EXEC all-1 for WMMA

    int laneHalf = lane >> 4;                  // 0 or 1
    int laneMod  = lane & 15;
    int row      = n0 + laneMod;
    int c0       = laneHalf * 8;               // K sub-offset within 32-wide window

    v8f acc[4] = {v8f{}, v8f{}, v8f{}, v8f{}};

#pragma unroll
    for (int kk = 0; kk < 4; ++kk) {
        // ---- streamed A fragment: 4x global_load_b128 + f32->f16 convert ----
        const float* src = (kk < 2 ? W : U) + (size_t)row * 64 + (kk & 1) * 32 + c0;
        v4f p0 = *(const v4f*)(src);
        v4f p1 = *(const v4f*)(src + 4);
        v4f p2 = *(const v4f*)(src + 16);
        v4f p3 = *(const v4f*)(src + 20);
        v16h a;
#pragma unroll
        for (int i = 0; i < 4; ++i) {
            a[i]      = (_Float16)p0[i];
            a[4 + i]  = (_Float16)p1[i];
            a[8 + i]  = (_Float16)p2[i];
            a[12 + i] = (_Float16)p3[i];
        }
        // ---- B fragments from 16KB L2-resident AT: 16 strided b16 loads each ----
        const _Float16* bp = AT + (size_t)(kk * 32 + c0) * NB + laneMod;
        v16h b0, b1, b2, b3;
#pragma unroll
        for (int i = 0; i < 8; ++i) {
            b0[i] = bp[i * NB];      b0[8 + i] = bp[(16 + i) * NB];
            b1[i] = bp[i * NB + 16]; b1[8 + i] = bp[(16 + i) * NB + 16];
            b2[i] = bp[i * NB + 32]; b2[8 + i] = bp[(16 + i) * NB + 32];
            b3[i] = bp[i * NB + 48]; b3[8 + i] = bp[(16 + i) * NB + 48];
        }

        acc[0] = __builtin_amdgcn_wmma_f32_16x16x32_f16(false, a, false, b0,
                                                        (short)0, acc[0], false, false);
        acc[1] = __builtin_amdgcn_wmma_f32_16x16x32_f16(false, a, false, b1,
                                                        (short)0, acc[1], false, false);
        acc[2] = __builtin_amdgcn_wmma_f32_16x16x32_f16(false, a, false, b2,
                                                        (short)0, acc[2], false, false);
        acc[3] = __builtin_amdgcn_wmma_f32_16x16x32_f16(false, a, false, b3,
                                                        (short)0, acc[3], false, false);
    }

    // ---- sigmoid + scatter store through np_indices ----
    int rbase = n0 + laneHalf * 8;
#pragma unroll
    for (int bt = 0; bt < 4; ++bt) {
        int bcol = bt * 16 + laneMod;
        float* vb = V + (size_t)bcol * n_atoms;
#pragma unroll
        for (int j = 0; j < 8; ++j) {
            int col = npidx[rbase + j];
            float x = acc[bt][j];
            vb[col] = 1.0f / (1.0f + __expf(-x));
        }
    }
}

// ---------------------------------------------------------------------------
// Kernel 4: background-knowledge atoms: V[b, bk[i]] += 1.0 (duplicates stack).
// ---------------------------------------------------------------------------
__global__ void bk_add_kernel(const int* __restrict__ bk, float* __restrict__ V,
                              int n_bk, int n_atoms) {
    int t = blockIdx.x * blockDim.x + threadIdx.x;
    if (t >= n_bk * NB) return;
    int i = t >> 6;        // index into bk
    int b = t & (NB - 1);  // batch row
    atomicAdd(V + (size_t)b * n_atoms + bk[i], 1.0f);
}

// ---------------------------------------------------------------------------
extern "C" void kernel_launch(void* const* d_in, const int* in_sizes, int n_in,
                              void* d_out, int out_size, void* d_ws, size_t ws_size,
                              hipStream_t stream) {
    const float* Z     = (const float*)d_in[0];
    const float* Q     = (const float*)d_in[1];
    const float* W     = (const float*)d_in[2];
    const float* U     = (const float*)d_in[3];
    const int*   npidx = (const int*)d_in[4];
    const int*   bk    = (const int*)d_in[5];

    int n_np    = in_sizes[4];
    int n_bk    = in_sizes[5];
    int B       = in_sizes[1] / QDIM;            // 64
    int E       = in_sizes[0] / (B * DD);        // 16
    int n_atoms = out_size / B;                  // 200000

    float*     V  = (float*)d_out;
    _Float16*  AT = (_Float16*)d_ws;             // 16 KB A-transpose staging

    // 1) reduce Z over E, pack [Zsum|Q] as f16 K-major
    prep_at_kernel<<<(KDIM * NB + 255) / 256, 256, 0, stream>>>(Z, Q, AT, E);

    // 2) zero the non-predicate tail of V
    if (n_atoms > n_np) {
        long total4 = (long)NB * ((n_atoms - n_np) >> 2);
        int blocks = (int)((total4 + 255) / 256);
        zero_tail_kernel<<<blocks, 256, 0, stream>>>(V, n_np, n_atoms);
    }

    // 3) WMMA GEMM + sigmoid, scores scattered into V[:, np_indices]
    int tiles  = (n_np + 15) / 16;               // 10000
    int blocks = (tiles + 3) / 4;                // 4 waves/block, 1 tile/wave
    gemm_sigmoid_kernel<<<blocks, 128, 0, stream>>>(W, U, AT, npidx, V, n_np, n_atoms);

    // 4) background knowledge atomic adds (after scores are in place)
    int total = n_bk * NB;
    bk_add_kernel<<<(total + 255) / 256, 256, 0, stream>>>(bk, V, n_bk, n_atoms);
}
